// MaskedGCN_28741921145401
// MI455X (gfx1250) — compile-verified, matrix operand order checked by reading, not run on hardware
//
#include <hip/hip_runtime.h>
#include <hip/hip_bf16.h>
#include <math.h>

#define GCN_N_NODES 100000
#define GCN_N_FEAT  500
#define GCN_N_HID   256
#define GCN_N_CLASS 40
#define GCN_N_EDGES 3200000

typedef __attribute__((ext_vector_type(16))) __bf16 v16bf;
typedef __attribute__((ext_vector_type(8)))  float  v8f;
typedef __attribute__((ext_vector_type(4)))  unsigned int v4u;
typedef __attribute__((ext_vector_type(8)))  unsigned int v8u;

// ---------- bf16 split helpers (manual RNE, bit-exact control) ----------
__device__ __forceinline__ unsigned short f32_to_bf16_rne(float x) {
    union { float f; unsigned int u; } v; v.f = x;
    unsigned int u = v.u;
    u += 0x7FFFu + ((u >> 16) & 1u);
    return (unsigned short)(u >> 16);
}
__device__ __forceinline__ float bf16_bits_to_f32(unsigned short h) {
    union { float f; unsigned int u; } v; v.u = ((unsigned int)h) << 16;
    return v.f;
}
__device__ __forceinline__ void split_bf16(float x, unsigned short& hi, unsigned short& lo) {
    hi = f32_to_bf16_rne(x);
    lo = f32_to_bf16_rne(x - bf16_bits_to_f32(hi));
}

// Fragment position (lane*16 + elem) for 16-bit 16x32 A/B WMMA operand tiles.
__device__ __forceinline__ int frag_pos(int mn, int k) {
    int lane = (((k >> 3) & 1) << 4) + mn;
    int e = ((k >> 4) << 3) | (k & 7);
    return lane * 16 + e;
}

__device__ __forceinline__ v8f wmma_bf16x3(v16bf ahi, v16bf alo, v16bf bhi, v16bf blo, v8f acc) {
    acc = __builtin_amdgcn_wmma_f32_16x16x32_bf16(false, ahi, false, bhi, (short)0, acc, false, false);
    acc = __builtin_amdgcn_wmma_f32_16x16x32_bf16(false, ahi, false, blo, (short)0, acc, false, false);
    acc = __builtin_amdgcn_wmma_f32_16x16x32_bf16(false, alo, false, bhi, (short)0, acc, false, false);
    return acc;
}

// ---------- Tensor Data Mover: 2D f32 tile, global -> LDS (D# built in SGPRs) ----------
// Loads tile_h rows x tile_w elems (f32) starting at gptr, row stride stride_w elems,
// contiguous into LDS at lds_byte_addr.  Issued per-wave; tracked by TENSORcnt.
__device__ __forceinline__ void tdm_load_2d_f32(
    unsigned int lds_byte_addr, const void* gptr,
    unsigned int tensor_w, unsigned int tensor_h,
    unsigned int tile_w, unsigned int tile_h, unsigned int stride_w)
{
    unsigned long long ga = (unsigned long long)gptr;
    v4u g0;
    g0.x = 1u;                                            // count=1, user desc
    g0.y = lds_byte_addr;                                 // lds_addr[63:32]
    g0.z = (unsigned int)(ga & 0xFFFFFFFFu);              // global_addr lo
    g0.w = ((unsigned int)(ga >> 32) & 0x01FFFFFFu)       // global_addr[56:32]
         | 0x80000000u;                                   // type=2 (image)
    v8u g1;
    g1.s0 = 2u << 16;                                     // data_size=4B, mask=0
    g1.s1 = (tensor_w & 0xFFFFu) << 16;                   // tensor_dim0 lo16
    g1.s2 = (tensor_w >> 16) | ((tensor_h & 0xFFFFu) << 16);
    g1.s3 = (tensor_h >> 16) | (tile_w << 16);            // tile_dim0
    g1.s4 = tile_h & 0xFFFFu;                             // tile_dim1 (tile_dim2=0)
    g1.s5 = stride_w;                                     // tensor_dim0_stride lo32
    g1.s6 = 0u;
    g1.s7 = 0u;
    v4u g2 = (v4u)0u;                                     // dims 2/3 unused
    v4u g3 = (v4u)0u;                                     // dim 4 unused
    asm volatile("tensor_load_to_lds %0, %1, %2, %3"
                 :: "s"(g0), "s"(g1), "s"(g2), "s"(g3)
                 : "memory");
}

// ---------------- GEMM1: h = features @ W1 + b1  [100000,500]x[500,256] ----------------
// grid = 6250 (16 rows each), block = 512 (16 waves, one 16x16 N-tile per wave)
__global__ __launch_bounds__(512) void gcn_gemm1(
    const float* __restrict__ A, const float* __restrict__ W,
    const float* __restrict__ b, float* __restrict__ C)
{
    __shared__ float          lrawB[32 * 256];                   // raw W1 tile (TDM dest)
    __shared__ unsigned short lAhi[16 * 32], lAlo[16 * 32];
    __shared__ unsigned short lBhi[16 * 512], lBlo[16 * 512];
    const int tid  = threadIdx.x;
    const int lane = tid & 31;
    const int w    = tid >> 5;          // wave id = N tile
    const int m0   = blockIdx.x * 16;
    const int n0   = w * 16;
    const unsigned int rawB_lds = (unsigned int)(unsigned long long)(void*)lrawB;

    v8f acc = (v8f)0.0f;
    for (int kk = 0; kk < 512; kk += 32) {
        // ---- TDM: DMA raw 32x256 f32 slab of W1 into LDS (wave 0 only) ----
        if (w == 0) {
            if (kk < GCN_N_FEAT) {
                unsigned int th = (unsigned int)min(32, GCN_N_FEAT - kk);
                tdm_load_2d_f32(rawB_lds, W + (size_t)kk * GCN_N_HID,
                                GCN_N_HID, GCN_N_FEAT, GCN_N_HID, th, GCN_N_HID);
                __builtin_amdgcn_s_wait_tensorcnt(0);
            }
        }
        // ---- stage A tile 16x32 (one element per thread), zero-pad K>=500 ----
        {
            int m = tid >> 5, k = tid & 31;
            int kg = kk + k;
            float x = (kg < GCN_N_FEAT) ? A[(size_t)(m0 + m) * GCN_N_FEAT + kg] : 0.0f;
            unsigned short hi, lo; split_bf16(x, hi, lo);
            int p = frag_pos(m, k);
            lAhi[p] = hi; lAlo[p] = lo;
        }
        if (kk + 32 < GCN_N_FEAT) {   // prefetch next A slab (global_prefetch_b8)
            __builtin_prefetch(&A[(size_t)(m0 + (tid >> 5)) * GCN_N_FEAT + kk + 32], 0, 1);
        }
        __syncthreads();              // publish TDM tile + A fragments
        // ---- convert raw B tile -> bf16 hi/lo fragments (float4 LDS reads) ----
        const float4* rawB4 = (const float4*)lrawB;
        #pragma unroll
        for (int i = 0; i < 4; ++i) {
            int q = i * 512 + tid;           // 2048 float4 = 32x256 f32
            int k = q >> 6;                  // 64 quads per row
            int n = (q & 63) << 2;
            int kg = kk + k;
            float4 x4 = make_float4(0.f, 0.f, 0.f, 0.f);
            if (kg < GCN_N_FEAT) x4 = rawB4[q];
            int p0 = (n >> 4) * 512 + frag_pos(n & 15, k);
            unsigned short hi, lo;
            split_bf16(x4.x, hi, lo); lBhi[p0     ] = hi; lBlo[p0     ] = lo;
            split_bf16(x4.y, hi, lo); lBhi[p0 + 16] = hi; lBlo[p0 + 16] = lo;
            split_bf16(x4.z, hi, lo); lBhi[p0 + 32] = hi; lBlo[p0 + 32] = lo;
            split_bf16(x4.w, hi, lo); lBhi[p0 + 48] = hi; lBlo[p0 + 48] = lo;
        }
        __syncthreads();
        v16bf ahi = *(const v16bf*)&lAhi[lane * 16];
        v16bf alo = *(const v16bf*)&lAlo[lane * 16];
        v16bf bhi = *(const v16bf*)&lBhi[w * 512 + lane * 16];
        v16bf blo = *(const v16bf*)&lBlo[w * 512 + lane * 16];
        acc = wmma_bf16x3(ahi, alo, bhi, blo, acc);
        __syncthreads();
    }
    // C/D layout: VGPR r -> row = r + 8*(lane>=16), col = lane&15
    const int col   = n0 + (lane & 15);
    const int rbase = m0 + ((lane >> 4) << 3);
    const float bias = b[col];
    #pragma unroll
    for (int r = 0; r < 8; ++r)
        C[(size_t)(rbase + r) * GCN_N_HID + col] = acc[r] + bias;
}

// ---------------- GEMM2: o = relu(h') @ W2 + b2  [100000,256]x[256,40->48] ----------------
// grid = ceil(100000/64), block = 384 (12 waves = 4 M-tiles x 3 N-tiles)
__global__ __launch_bounds__(384) void gcn_gemm2(
    const float* __restrict__ H, const float* __restrict__ W,
    const float* __restrict__ b, float* __restrict__ C)
{
    __shared__ unsigned short lWhi[8 * 3 * 512], lWlo[8 * 3 * 512];  // all of padded W2
    __shared__ unsigned short lAhi[4 * 512],     lAlo[4 * 512];
    const int tid  = threadIdx.x;
    const int lane = tid & 31;
    const int w    = tid >> 5;      // 0..11
    const int mt   = w / 3, nt = w % 3;
    const int mblk = blockIdx.x * 64;
    const int m0   = mblk + mt * 16;
    const int n0   = nt * 16;

    // stage entire 256x48 (zero-padded) W2 once, fragment order per (kchunk, ntile)
    #pragma unroll 4
    for (int i = 0; i < 32; ++i) {
        int idx = i * 384 + tid;         // < 12288
        int k = idx / 48, n = idx % 48;
        float x = (n < GCN_N_CLASS) ? W[k * GCN_N_CLASS + n] : 0.0f;
        unsigned short hi, lo; split_bf16(x, hi, lo);
        int p = ((k >> 5) * 3 + (n >> 4)) * 512 + frag_pos(n & 15, k & 31);
        lWhi[p] = hi; lWlo[p] = lo;
    }
    __syncthreads();

    v8f acc = (v8f)0.0f;
    for (int kc = 0; kc < 8; ++kc) {
        for (int idx = tid; idx < 2048; idx += 384) {   // A tile 64x32, relu fused
            int m = idx >> 5, k = idx & 31;
            int row = mblk + m;
            float x = 0.0f;
            if (row < GCN_N_NODES) {
                x = H[(size_t)row * GCN_N_HID + kc * 32 + k];
                x = x > 0.0f ? x : 0.0f;
            }
            unsigned short hi, lo; split_bf16(x, hi, lo);
            int p = (m >> 4) * 512 + frag_pos(m & 15, k);
            lAhi[p] = hi; lAlo[p] = lo;
        }
        __syncthreads();
        v16bf ahi = *(const v16bf*)&lAhi[mt * 512 + lane * 16];
        v16bf alo = *(const v16bf*)&lAlo[mt * 512 + lane * 16];
        v16bf bhi = *(const v16bf*)&lWhi[(kc * 3 + nt) * 512 + lane * 16];
        v16bf blo = *(const v16bf*)&lWlo[(kc * 3 + nt) * 512 + lane * 16];
        acc = wmma_bf16x3(ahi, alo, bhi, blo, acc);
        __syncthreads();
    }
    const int col   = n0 + (lane & 15);
    const int rbase = ((lane >> 4) << 3);
    if (col < GCN_N_CLASS) {
        const float bias = b[col];
        #pragma unroll
        for (int r = 0; r < 8; ++r) {
            int row = m0 + rbase + r;
            if (row < GCN_N_NODES)
                C[(size_t)row * GCN_N_CLASS + col] = acc[r] + bias;
        }
    }
}

// ---------------- SpMM (hidden, 256 feats): one wave32 per edge ----------------
__global__ __launch_bounds__(256) void gcn_spmm_hid(
    const int* __restrict__ er, const int* __restrict__ ec, const float* __restrict__ ev,
    const float* __restrict__ X, float* __restrict__ Y)
{
    const int e    = blockIdx.x * 8 + (threadIdx.x >> 5);
    const int lane = threadIdx.x & 31;
    const int r = er[e], c = ec[e];
    const float v = ev[e];
    const float4* xp = (const float4*)(X + (size_t)c * GCN_N_HID);
    float4 x0 = xp[lane * 2 + 0];
    float4 x1 = xp[lane * 2 + 1];
    float* yp = Y + (size_t)r * GCN_N_HID + lane * 8;
    unsafeAtomicAdd(yp + 0, v * x0.x); unsafeAtomicAdd(yp + 1, v * x0.y);
    unsafeAtomicAdd(yp + 2, v * x0.z); unsafeAtomicAdd(yp + 3, v * x0.w);
    unsafeAtomicAdd(yp + 4, v * x1.x); unsafeAtomicAdd(yp + 5, v * x1.y);
    unsafeAtomicAdd(yp + 6, v * x1.z); unsafeAtomicAdd(yp + 7, v * x1.w);
}

// ---------------- SpMM (class, 40 feats): one wave32 per edge ----------------
__global__ __launch_bounds__(256) void gcn_spmm_cls(
    const int* __restrict__ er, const int* __restrict__ ec, const float* __restrict__ ev,
    const float* __restrict__ X, float* __restrict__ Y)
{
    const int e    = blockIdx.x * 8 + (threadIdx.x >> 5);
    const int lane = threadIdx.x & 31;
    const int r = er[e], c = ec[e];
    const float v = ev[e];
    const float* xp = X + (size_t)c * GCN_N_CLASS;
    float* yp = Y + (size_t)r * GCN_N_CLASS;
    unsafeAtomicAdd(yp + lane, v * xp[lane]);
    if (lane < 8) unsafeAtomicAdd(yp + 32 + lane, v * xp[32 + lane]);
}

// ---------------- zero init ----------------
__global__ void gcn_zero(float* __restrict__ p, long n) {
    long i = (long)blockIdx.x * blockDim.x + threadIdx.x;
    if (i < n) p[i] = 0.0f;
}

// ---------------- row-wise log_softmax over 40 classes: one wave per node ----------------
__global__ __launch_bounds__(256) void gcn_logsoftmax(
    const float* __restrict__ X, float* __restrict__ O)
{
    const int node = blockIdx.x * 8 + (threadIdx.x >> 5);
    const int lane = threadIdx.x & 31;
    if (node >= GCN_N_NODES) return;
    const float* xp = X + (size_t)node * GCN_N_CLASS;
    float v0 = xp[lane];
    float v1 = (lane < 8) ? xp[32 + lane] : -INFINITY;
    float m = fmaxf(v0, v1);
    #pragma unroll
    for (int off = 16; off > 0; off >>= 1) m = fmaxf(m, __shfl_xor(m, off, 32));
    float s = expf(v0 - m) + ((lane < 8) ? expf(v1 - m) : 0.0f);
    #pragma unroll
    for (int off = 16; off > 0; off >>= 1) s += __shfl_xor(s, off, 32);
    float ls = m + logf(s);
    O[(size_t)node * GCN_N_CLASS + lane] = v0 - ls;
    if (lane < 8) O[(size_t)node * GCN_N_CLASS + 32 + lane] = v1 - ls;
}

extern "C" void kernel_launch(void* const* d_in, const int* in_sizes, int n_in,
                              void* d_out, int out_size, void* d_ws, size_t ws_size,
                              hipStream_t stream) {
    const float* features = (const float*)d_in[0];
    const int*   edge_row = (const int*)d_in[1];
    const int*   edge_col = (const int*)d_in[2];
    const float* edge_val = (const float*)d_in[3];
    const float* W1       = (const float*)d_in[4];
    const float* b1       = (const float*)d_in[5];
    const float* W2       = (const float*)d_in[6];
    const float* b2       = (const float*)d_in[7];
    float*       out      = (float*)d_out;

    float* ws = (float*)d_ws;
    float* h  = ws;                                      // [N, 256]
    float* h2 = ws + (size_t)GCN_N_NODES * GCN_N_HID;    // [N, 256]
    float* o  = ws;                                      // [N, 40]  (reuses dead h)
    float* o2 = ws + (size_t)GCN_N_NODES * GCN_N_CLASS;  // [N, 40]  (also inside dead h)

    // layer 1 linear
    gcn_gemm1<<<GCN_N_NODES / 16, 512, 0, stream>>>(features, W1, b1, h);
    // zero spmm target, then spmm
    {
        long n = (long)GCN_N_NODES * GCN_N_HID;
        gcn_zero<<<(unsigned)((n + 255) / 256), 256, 0, stream>>>(h2, n);
    }
    gcn_spmm_hid<<<GCN_N_EDGES / 8, 256, 0, stream>>>(edge_row, edge_col, edge_val, h, h2);
    // layer 2 linear (relu fused on load of h2)
    gcn_gemm2<<<(GCN_N_NODES + 63) / 64, 384, 0, stream>>>(h2, W2, b2, o);
    {
        long n = (long)GCN_N_NODES * GCN_N_CLASS;
        gcn_zero<<<(unsigned)((n + 255) / 256), 256, 0, stream>>>(o2, n);
    }
    gcn_spmm_cls<<<GCN_N_EDGES / 8, 256, 0, stream>>>(edge_row, edge_col, edge_val, o, o2);
    gcn_logsoftmax<<<(GCN_N_NODES + 7) / 8, 256, 0, stream>>>(o2, out);
}
